// SOMLayer_35132832481413
// MI455X (gfx1250) — matrix operand order repaired and checked by reading, not compiled, not run on hardware
//
#include <hip/hip_runtime.h>
#include <hip/hip_bf16.h>

// ---------------------------------------------------------------------------
// SOM step for MI455X (gfx1250): bf16 WMMA for both big GEMMs, f32 epilogues.
// ---------------------------------------------------------------------------

typedef __attribute__((ext_vector_type(16))) __bf16 v16bf;
typedef __attribute__((ext_vector_type(8)))  __bf16 v8bf;
typedef __attribute__((ext_vector_type(4)))  __bf16 v4bf;
typedef __attribute__((ext_vector_type(8)))  float  v8f;

static constexpr int B_SAMP = 1024;
static constexpr int D_DIM  = 128;
static constexpr int W_GRID = 256;   // H=W=256, N=65536
static constexpr int N_NEUR = 65536;

__device__ __forceinline__ unsigned long long pack_di(float d, unsigned int idx) {
  return ((unsigned long long)__float_as_uint(d) << 32) | (unsigned long long)idx;
}

__device__ __forceinline__ unsigned long long shfl_xor_u64_w16(unsigned long long v, int m) {
  unsigned int lo = (unsigned int)v;
  unsigned int hi = (unsigned int)(v >> 32);
  lo = __shfl_xor(lo, m, 16);
  hi = __shfl_xor(hi, m, 16);
  return ((unsigned long long)hi << 32) | (unsigned long long)lo;
}

__device__ __forceinline__ v16bf frag_from(const __bf16* p0, const __bf16* p1) {
  v16bf r;
  ((v8bf*)&r)[0] = *(const v8bf*)p0;
  ((v8bf*)&r)[1] = *(const v8bf*)p1;
  return r;
}

// ---------------------------------------------------------------------------
// Kernel 0: convert inputs f32->bf16 into ws; init packed argmin array.
// ---------------------------------------------------------------------------
__global__ __launch_bounds__(256) void som_prep_kernel(
    const float* __restrict__ x, __bf16* __restrict__ xbf,
    unsigned long long* __restrict__ gmin) {
  int id = blockIdx.x * 256 + threadIdx.x;
  if (id < B_SAMP * D_DIM) xbf[id] = (__bf16)x[id];
  if (id < B_SAMP) gmin[id] = ~0ULL;
}

// ---------------------------------------------------------------------------
// Kernel 1: BMU search.  Grid = 512 WGs (128 neurons each), 8 waves.
// d2 tile = xnorm + wnorm - 2 * (X @ W^T) via v_wmma_f32_16x16x32_bf16.
// ---------------------------------------------------------------------------
__global__ __launch_bounds__(256) void som_bmu_kernel(
    const __bf16* __restrict__ xbf, const float* __restrict__ wsrc,
    unsigned long long* __restrict__ gmin) {
  constexpr int KS = 136;  // padded LDS row stride (bf16 elems), 272B: 16B aligned
  __shared__ alignas(16) __bf16 lds_w[128 * KS];
  __shared__ alignas(16) __bf16 lds_x[32 * KS];
  __shared__ float wnorm_s[128];
  __shared__ float xnorm_s[32];
  __shared__ unsigned long long smin[32];

  const int t    = threadIdx.x;
  const int wave = t >> 5;
  const int lane = t & 31;
  const int ln   = lane & 15;
  const int hi   = lane >> 4;
  const int nbase = blockIdx.x * 128;  // global neuron base of this tile

  // ---- stage W tile (128 neurons x 128 dims), f32 -> bf16, row-major [n][k]
  {
    const float* wp = wsrc + (size_t)nbase * D_DIM;
    const int n  = t >> 1;
    const int k0 = (t & 1) * 64;
#pragma unroll
    for (int j = 0; j < 16; ++j) {
      float4 f = *(const float4*)(wp + (size_t)n * D_DIM + k0 + j * 4);
      v4bf b;
      b[0] = (__bf16)f.x; b[1] = (__bf16)f.y; b[2] = (__bf16)f.z; b[3] = (__bf16)f.w;
      *(v4bf*)&lds_w[n * KS + k0 + j * 4] = b;
    }
  }
  __syncthreads();
  if (t < 128) {
    float s = 0.f;
#pragma unroll 8
    for (int k = 0; k < D_DIM; ++k) { float v = (float)lds_w[t * KS + k]; s += v * v; }
    wnorm_s[t] = s;
  }

  const int sb  = wave >> 2;        // sample block (0..1) of 16
  const int nbp = (wave & 3) * 2;   // first of 2 neuron blocks for this wave

  for (int tile = 0; tile < 32; ++tile) {
    // ---- stage X tile (32 samples x 128), bf16 copy from ws
    {
      const int row = t >> 3;
      const int off = (t & 7) * 16;
      const __bf16* src = xbf + (size_t)(tile * 32 + row) * D_DIM + off;
      v8bf a0 = *(const v8bf*)src;
      v8bf a1 = *(const v8bf*)(src + 8);
      *(v8bf*)&lds_x[row * KS + off]     = a0;
      *(v8bf*)&lds_x[row * KS + off + 8] = a1;
    }
    if (t < 32) smin[t] = ~0ULL;
    __syncthreads();
    if (t < 32) {
      float s = 0.f;
#pragma unroll 8
      for (int k = 0; k < D_DIM; ++k) { float v = (float)lds_x[t * KS + k]; s += v * v; }
      xnorm_s[t] = s;
    }
    __syncthreads();

    // ---- WMMA: 2 sample blocks x 8 neuron blocks, each wave owns 1 x 2
    const v8f vzero = {0.f, 0.f, 0.f, 0.f, 0.f, 0.f, 0.f, 0.f};
    v8f c0 = vzero, c1 = vzero;
#pragma unroll
    for (int kb = 0; kb < D_DIM; kb += 32) {
      const __bf16* xr = &lds_x[(sb * 16 + ln) * KS + kb];
      v16bf a = frag_from(xr + 16 * hi, xr + 16 * hi + 8);
      const __bf16* w0 = &lds_w[(nbp * 16 + ln) * KS + kb + 16 * hi];
      const __bf16* w1 = &lds_w[((nbp + 1) * 16 + ln) * KS + kb + 16 * hi];
      v16bf b0 = frag_from(w0, w0 + 8);
      v16bf b1 = frag_from(w1, w1 + 8);
      c0 = __builtin_amdgcn_wmma_f32_16x16x32_bf16(false, a, false, b0, (short)0, c0, false, false);
      c1 = __builtin_amdgcn_wmma_f32_16x16x32_bf16(false, a, false, b1, (short)0, c1, false, false);
    }

    // ---- d2 + per-row argmin (butterfly in 16-lane groups), combine in LDS
#pragma unroll
    for (int v = 0; v < 8; ++v) {
      const int sl = sb * 16 + v + 8 * hi;  // sample-in-tile for this C row
      const float xn = xnorm_s[sl];
      float d0 = fmaxf(xn + wnorm_s[nbp * 16 + ln] - 2.f * c0[v], 0.f);
      float d1 = fmaxf(xn + wnorm_s[(nbp + 1) * 16 + ln] - 2.f * c1[v], 0.f);
      unsigned long long p0 = pack_di(d0, (unsigned)(nbase + nbp * 16 + ln));
      unsigned long long p1 = pack_di(d1, (unsigned)(nbase + (nbp + 1) * 16 + ln));
      unsigned long long p = p0 < p1 ? p0 : p1;
#pragma unroll
      for (int off = 8; off; off >>= 1) {
        unsigned long long q = shfl_xor_u64_w16(p, off);
        p = p < q ? p : q;
      }
      if (ln == 0) atomicMin(&smin[sl], p);  // ds_min_u64
    }
    __syncthreads();
    if (t < 32) atomicMin(&gmin[tile * 32 + t], smin[t]);  // global_atomic_min_u64
    if (tile + 1 < 32)
      __builtin_prefetch(xbf + (size_t)(tile + 1) * 32 * D_DIM + t, 0, 0);
    __syncthreads();
  }
}

// ---------------------------------------------------------------------------
// Kernel 2: weight update.  Grid = 1024 WGs (64 neurons x 128 dims each),
// loop over all 1024 samples in chunks of 32 (= WMMA K).
// num[n][d] += h[n][s] * x[s][d]   via v_wmma_f32_16x16x32_bf16.
// ---------------------------------------------------------------------------
__global__ __launch_bounds__(256) void som_update_kernel(
    const __bf16* __restrict__ xbf, const unsigned long long* __restrict__ gmin,
    const float* __restrict__ wsrc, float* __restrict__ out_w) {
  constexpr int HS = 40;  // 80B stride, 16B aligned
  constexpr int XS = 40;
  __shared__ alignas(16) __bf16 lds_h[64 * HS];    // [neuron][sample]
  __shared__ alignas(16) __bf16 lds_xt[128 * XS];  // [dim][sample] (transposed)
  __shared__ float bi_s[32], bj_s[32], hpart[64], hsum_s[64];

  const int t    = threadIdx.x;
  const int wave = t >> 5;
  const int lane = t & 31;
  const int ln   = lane & 15;
  const int hi   = lane >> 4;
  const int nt   = blockIdx.x * 64;      // neuron tile base
  const int r    = wave >> 1;            // neuron row block (0..3)
  const int cbase = (wave & 1) * 4;      // first of 4 dim blocks

  // h-compute assignment: 4 threads per neuron row, 8 samples each
  const int hn   = t >> 2;
  const int soff = (t & 3) * 8;
  const float ni = (float)((nt + hn) >> 8);
  const float nj = (float)((nt + hn) & (W_GRID - 1));

  const v8f vzero = {0.f, 0.f, 0.f, 0.f, 0.f, 0.f, 0.f, 0.f};
  v8f acc[4] = {vzero, vzero, vzero, vzero};
  float hsum = 0.f;

  for (int ch = 0; ch < 32; ++ch) {
    const int sb0 = ch * 32;
    if (t < 32) {
      unsigned int idx = (unsigned int)gmin[sb0 + t];
      bi_s[t] = (float)(idx >> 8);
      bj_s[t] = (float)(idx & (W_GRID - 1));
    }
    if (t < 64) hpart[t] = 0.f;
    // ---- stage X chunk transposed: [d][s]
    {
      const int s    = t >> 3;
      const int doff = (t & 7) * 16;
      const __bf16* src = xbf + (size_t)(sb0 + s) * D_DIM + doff;
      v8bf a0 = *(const v8bf*)src;
      v8bf a1 = *(const v8bf*)(src + 8);
#pragma unroll
      for (int i = 0; i < 8; ++i) lds_xt[(doff + i) * XS + s] = a0[i];
#pragma unroll
      for (int i = 0; i < 8; ++i) lds_xt[(doff + 8 + i) * XS + s] = a1[i];
    }
    __syncthreads();

    // ---- gaussian neighborhood h (v_exp_f32) + partial h_sum
    {
      float part = 0.f;
#pragma unroll
      for (int s8 = 0; s8 < 8; ++s8) {
        const int s = soff + s8;
        float di = ni - bi_s[s];
        float dj = nj - bj_s[s];
        float h  = __expf(-(di * di + dj * dj) * 0.125f);  // 1/(2*sigma^2), sigma=2
        part += h;
        lds_h[hn * HS + s] = (__bf16)h;
      }
      atomicAdd(&hpart[hn], part);  // ds_add_f32
    }
    __syncthreads();
    if (t < 64) hsum += hpart[t];

    // ---- WMMA: each wave 1 A-frag + 4 B-frags + 4 WMMAs
    const __bf16* hr = &lds_h[(r * 16 + ln) * HS];
    v16bf a = frag_from(hr + 16 * hi, hr + 16 * hi + 8);
#pragma unroll
    for (int cb = 0; cb < 4; ++cb) {
      const __bf16* xp = &lds_xt[((cbase + cb) * 16 + ln) * XS + 16 * hi];
      v16bf b = frag_from(xp, xp + 8);
      acc[cb] = __builtin_amdgcn_wmma_f32_16x16x32_bf16(false, a, false, b, (short)0,
                                                        acc[cb], false, false);
    }
    __syncthreads();
  }

  if (t < 64) hsum_s[t] = hsum;
  __syncthreads();

  // ---- fused elementwise update:  w + lr*(num - h_sum*w)/B
  const float scale = 0.1f / (float)B_SAMP;  // lr = 0.1*(1 - 0/1000)
#pragma unroll
  for (int cb = 0; cb < 4; ++cb) {
    const int d = (cbase + cb) * 16 + ln;
#pragma unroll
    for (int v = 0; v < 8; ++v) {
      const int nl = r * 16 + v + 8 * hi;
      const size_t g = (size_t)(nt + nl) * D_DIM + d;
      const float wo = wsrc[g];
      out_w[g] = wo + scale * (acc[cb][v] - hsum_s[nl] * wo);
    }
  }
}

// ---------------------------------------------------------------------------
// Kernel 3: bmu coords + exact f32 quantization error.
// ---------------------------------------------------------------------------
__global__ __launch_bounds__(128) void som_out_kernel(
    const float* __restrict__ x, const float* __restrict__ w,
    const unsigned long long* __restrict__ gmin, float* __restrict__ out) {
  const int b = blockIdx.x * blockDim.x + threadIdx.x;
  if (b >= B_SAMP) return;
  const unsigned int idx = (unsigned int)gmin[b];
  out[2 * b]     = (float)(idx >> 8);
  out[2 * b + 1] = (float)(idx & (W_GRID - 1));
  const float* xr = x + (size_t)b * D_DIM;
  const float* wr = w + (size_t)idx * D_DIM;
  float s = 0.f;
#pragma unroll 4
  for (int k = 0; k < D_DIM; k += 4) {
    float4 xv = *(const float4*)(xr + k);
    float4 wv = *(const float4*)(wr + k);
    float d0 = xv.x - wv.x, d1 = xv.y - wv.y, d2 = xv.z - wv.z, d3 = xv.w - wv.w;
    s += d0 * d0 + d1 * d1 + d2 * d2 + d3 * d3;
  }
  out[2 * B_SAMP + b] = sqrtf(fmaxf(s, 0.f));
}

// ---------------------------------------------------------------------------
// d_out layout: bmu (1024x2) | q_err (1024) | new_weights (256*256*128)
// d_ws  layout: gmin (1024 u64 = 8KB) | xbf (1024*128 bf16 = 256KB)
// ---------------------------------------------------------------------------
extern "C" void kernel_launch(void* const* d_in, const int* in_sizes, int n_in,
                              void* d_out, int out_size, void* d_ws, size_t ws_size,
                              hipStream_t stream) {
  (void)in_sizes; (void)n_in; (void)out_size; (void)ws_size;
  const float* x = (const float*)d_in[0];
  const float* w = (const float*)d_in[1];
  float* out = (float*)d_out;

  unsigned long long* gmin = (unsigned long long*)d_ws;
  __bf16* xbf = (__bf16*)((char*)d_ws + 8192);

  som_prep_kernel<<<512, 256, 0, stream>>>(x, xbf, gmin);
  som_bmu_kernel<<<N_NEUR / 128, 256, 0, stream>>>(xbf, w, gmin);
  som_update_kernel<<<N_NEUR / 64, 256, 0, stream>>>(xbf, gmin, w, out + 3 * B_SAMP);
  som_out_kernel<<<B_SAMP / 128, 128, 0, stream>>>(x, w, gmin, out);
}
